// WindAwareEarthSpecificBlock_74277164417314
// MI455X (gfx1250) — matrix operand order repaired and checked by reading, not compiled, not run on hardware
//
#include <hip/hip_runtime.h>
#include <hip/hip_bf16.h>
#include <math.h>

// ---------------- static config ----------------
#define PL 8
#define LAT 48
#define LON 96
#define WP 2
#define WL 6
#define WW 12
#define SP 1
#define SL 3
#define SN 6
#define NH 16
#define HD 32
#define C 512
#define NTOK (PL*LAT*LON)          // 36864
#define NWIN 256                   // (LON/WW) * (PL/WP)*(LAT/WL) = 8*32
#define NW 32                      // windows per lon-slice
#define NTW 144                    // tokens per window

typedef __attribute__((ext_vector_type(16))) __bf16 v16bf;
typedef __attribute__((ext_vector_type(8)))  __bf16 v8bf;
typedef __attribute__((ext_vector_type(8)))  float  v8f;
typedef __attribute__((ext_vector_type(4)))  int    v4i;

#if __has_builtin(__builtin_amdgcn_global_load_async_to_lds_b128) && \
    __has_builtin(__builtin_amdgcn_s_wait_asynccnt)
#define HAVE_ASYNC_LDS 1
typedef __attribute__((address_space(1))) v4i GV4;   // global 128-bit chunk
typedef __attribute__((address_space(3))) v4i LV4;   // LDS 128-bit chunk
#endif

__device__ __forceinline__ __bf16 f2bf(float f) {
    union { float f; unsigned u; } x; x.f = f;
    unsigned r = x.u + 0x7FFFu + ((x.u >> 16) & 1u);
    union { unsigned short s; __bf16 b; } y; y.s = (unsigned short)(r >> 16);
    return y.b;
}

__device__ __forceinline__ v8f wmma_bf16(v16bf a, v16bf b, v8f c) {
    return __builtin_amdgcn_wmma_f32_16x16x32_bf16(false, a, false, b, (short)0, c, false, false);
}

// A fragment (16x32 bf16) from row-major [rows x ld]; rows at row0, K at k0.
// Contiguous K order (k = 16*half + s), matching load_b_frag: two adjacent 16B chunks per lane.
__device__ __forceinline__ v16bf load_a_frag(const __bf16* A, int ld, int row0, int k0, int lane) {
    int m = lane & 15, g = lane >> 4;
    const __bf16* p = A + (size_t)(row0 + m) * ld + k0 + g * 16;
    union { v16bf v; v8bf h[2]; } u;
    u.h[0] = *(const v8bf*)(p);
    u.h[1] = *(const v8bf*)(p + 8);
    return u.v;
}

// B fragment (32x16 bf16) where column n of B is a contiguous row of a
// row-major matrix M[n][k] with leading dim ldk.
__device__ __forceinline__ v16bf load_b_frag(const __bf16* Mt, int ldk, int n0, int k0, int lane) {
    int n = lane & 15, g = lane >> 4;
    const __bf16* p = Mt + (size_t)(n0 + n) * ldk + k0 + g * 16;
    return *(const v16bf*)(p);
}

// windowed row index -> natural token index (reverse partition + roll back)
__device__ __forceinline__ int win_to_nat(int r) {
    int bl = r / (NW * NTW);
    int rem = r - bl * (NW * NTW);
    int w = rem / NTW;
    int n = rem - w * NTW;
    int ip = w >> 3, il = w & 7;               // nPl=4, nLat=8
    int zp = n / (WL * WW); int n2 = n - zp * (WL * WW);
    int zl = n2 / WW, zw = n2 - zl * WW;
    int pl = ip * WP + zp + SP;  if (pl >= PL)  pl -= PL;
    int lat = il * WL + zl + SL; if (lat >= LAT) lat -= LAT;
    int lon = bl * WW + zw + SN; if (lon >= LON) lon -= LON;
    return (pl * LAT + lat) * LON + lon;
}

// natural token index -> windowed row index (roll + partition)
__device__ __forceinline__ int nat_to_win(int t) {
    int pl = t / (LAT * LON); int rem = t - pl * (LAT * LON);
    int lat = rem / LON, lon = rem - lat * LON;
    int plr = pl - SP;  if (plr < 0)  plr += PL;
    int latr = lat - SL; if (latr < 0) latr += LAT;
    int lonr = lon - SN; if (lonr < 0) lonr += LON;
    int ip = plr / WP, zp = plr - ip * WP;
    int il = latr / WL, zl = latr - il * WL;
    int bl = lonr / WW, zw = lonr - bl * WW;
    int w = ip * 8 + il;
    int n = (zp * WL + zl) * WW + zw;
    return (bl * NW + w) * NTW + n;
}

// ---------------- weight convert + transpose: W[K][N] f32 -> Wt[N][K] bf16 ----
__global__ __launch_bounds__(256)
void cvt_t_k(const float* __restrict__ W, __bf16* __restrict__ Wt, int K, int N) {
    int i = blockIdx.x * 256 + threadIdx.x;
    if (i < K * N) {
        int k = i / N, n = i - k * N;
        Wt[(size_t)n * K + k] = f2bf(W[i]);
    }
}

// ---------------- LayerNorm (optionally writing to windowed layout) ---------
__global__ __launch_bounds__(256)
void ln_k(const float* __restrict__ x, const float* __restrict__ gw,
          const float* __restrict__ gb, __bf16* __restrict__ out, int windowed) {
    int t = blockIdx.x;
    __shared__ float red[256];
    const float* xp = x + (size_t)t * C;
    float vals[2], s = 0.f, s2 = 0.f;
    for (int i = 0; i < 2; ++i) {
        float v = xp[threadIdx.x + 256 * i];
        vals[i] = v; s += v; s2 += v * v;
    }
    red[threadIdx.x] = s; __syncthreads();
    for (int o = 128; o; o >>= 1) { if (threadIdx.x < o) red[threadIdx.x] += red[threadIdx.x + o]; __syncthreads(); }
    float mean = red[0] * (1.0f / C); __syncthreads();
    red[threadIdx.x] = s2; __syncthreads();
    for (int o = 128; o; o >>= 1) { if (threadIdx.x < o) red[threadIdx.x] += red[threadIdx.x + o]; __syncthreads(); }
    float var = red[0] * (1.0f / C) - mean * mean;
    float inv = rsqrtf(var + 1e-5f);
    int r = windowed ? nat_to_win(t) : t;
    for (int i = 0; i < 2; ++i) {
        int c = threadIdx.x + 256 * i;
        out[(size_t)r * C + c] = f2bf((vals[i] - mean) * inv * gw[c] + gb[c]);
    }
}

// ---------------- WMMA GEMM --------------------------------------------------
// Block tile 128(M) x 256(N), 8 waves in 2x4 grid, 64x64 per wave
// (16 accumulators, 4x fragment reuse).  A tile double-buffered through LDS
// via the gfx1250 async-to-LDS path: tile i+1 loads overlap tile i's WMMAs;
// ASYNCcnt ordering guarantees wait(2) retires the current tile's pair.
// EPI 0: QKV (bf16 out, scale q part)   EPI 1: proj (+res, natural order, f32)
// EPI 2: MLP1 (bf16 out, exact GELU)    EPI 3: MLP2 (+res, f32 out)
template<int EPI>
__global__ __launch_bounds__(256)
void gemm_k(const __bf16* __restrict__ A, const __bf16* __restrict__ Wt,
            const float* __restrict__ bias, void* __restrict__ out,
            const float* __restrict__ res, int M, int N, int K) {
    int lane = threadIdx.x & 31;
    int wave = threadIdx.x >> 5;
    int wm = wave >> 2, wn = wave & 3;        // 2 x 4 wave grid
    int bm = blockIdx.y * 128 + wm * 64;
    int bn = blockIdx.x * 256 + wn * 64;

    __shared__ __bf16 As[2][128 * 32];        // double-buffered A tile (2 x 8 KB)
    int sr = threadIdx.x >> 1;                // 0..127: row in tile
    int sc = (threadIdx.x & 1) * 16;          // 0 or 16: K-chunk
    const __bf16* gsrc = A + (size_t)(blockIdx.y * 128 + sr) * K + sc;

    auto issue = [&](int k0, int buf) {
        __bf16* d = &As[buf][sr * 32 + sc];
#ifdef HAVE_ASYNC_LDS
        __builtin_amdgcn_global_load_async_to_lds_b128((GV4*)(gsrc + k0),     (LV4*)d,       0, 0);
        __builtin_amdgcn_global_load_async_to_lds_b128((GV4*)(gsrc + k0 + 8), (LV4*)(d + 8), 0, 0);
#else
        *(v8bf*)d       = *(const v8bf*)(gsrc + k0);
        *(v8bf*)(d + 8) = *(const v8bf*)(gsrc + k0 + 8);
#endif
    };

    v8f acc[4][4] = {};
    issue(0, 0);
    for (int k0 = 0, it = 0; k0 < K; k0 += 32, ++it) {
        int cur = it & 1;
        if (k0 + 32 < K) {
            issue(k0 + 32, cur ^ 1);          // prefetch next tile into other buffer
#ifdef HAVE_ASYNC_LDS
            __builtin_amdgcn_s_wait_asynccnt(2);   // retire current tile's pair
#endif
        } else {
#ifdef HAVE_ASYNC_LDS
            __builtin_amdgcn_s_wait_asynccnt(0);
#endif
        }
        __syncthreads();                      // all waves' chunks landed

        if (k0 + 32 < K) __builtin_prefetch(Wt + (size_t)bn * K + k0 + 32, 0, 3);

        v16bf af[4], bf[4];
        for (int mi = 0; mi < 4; ++mi) af[mi] = load_a_frag(As[cur], 32, wm * 64 + mi * 16, 0, lane);
        for (int ni = 0; ni < 4; ++ni) bf[ni] = load_b_frag(Wt, K, bn + ni * 16, k0, lane);
        for (int mi = 0; mi < 4; ++mi)
            for (int ni = 0; ni < 4; ++ni)
                acc[mi][ni] = wmma_bf16(af[mi], bf[ni], acc[mi][ni]);
        __syncthreads();                      // reads done before buffer reuse
    }

    int g = lane >> 4, nl = lane & 15;
    float bv[4];
    for (int ni = 0; ni < 4; ++ni) bv[ni] = bias[bn + ni * 16 + nl];
    for (int mi = 0; mi < 4; ++mi)
        for (int v = 0; v < 8; ++v) {
            int row = bm + mi * 16 + v + 8 * g;
            int t = (EPI == 1) ? win_to_nat(row) : row;
            for (int ni = 0; ni < 4; ++ni) {
                int col = bn + ni * 16 + nl;
                float val = acc[mi][ni][v] + bv[ni];
                if (EPI == 0) {
                    if (col < C) val *= 0.17677669529663689f;  // HD^-0.5
                    ((__bf16*)out)[(size_t)row * N + col] = f2bf(val);
                } else if (EPI == 1) {
                    ((float*)out)[(size_t)t * C + col] = res[(size_t)t * C + col] + val;
                } else if (EPI == 2) {
                    float gl = 0.5f * val * (1.0f + erff(val * 0.70710678118654752f));
                    ((__bf16*)out)[(size_t)row * N + col] = f2bf(gl);
                } else {
                    ((float*)out)[(size_t)row * C + col] = res[(size_t)row * C + col] + val;
                }
            }
        }
}

// ---------------- windowed attention: one block per (window, head) ---------
__global__ __launch_bounds__(288)
void attn_k(const __bf16* __restrict__ qkv, const float* __restrict__ bias_table,
            __bf16* __restrict__ o) {
    int u = blockIdx.x;          // window 0..255 (= bl*32 + w)
    int hh = blockIdx.y;         // head
    int tid = threadIdx.x;
    int wave = tid >> 5, lane = tid & 31;
    int g = lane >> 4, nl = lane & 15;

    __shared__ __bf16 vT[32 * 160];          // v transposed [d][row], K padded to 160
    __shared__ __bf16 Ps[9][16 * 160];       // per-wave softmax strip, padded

    const __bf16* qbase = qkv + (size_t)(u * NTW) * (3 * C) + hh * HD;
    const __bf16* kbase = qbase + C;
    const __bf16* vbase = qbase + 2 * C;

    // stage v transposed into LDS
    for (int idx = tid; idx < NTW * HD; idx += 288) {
        int row = idx >> 5, d = idx & 31;
        vT[d * 160 + row] = vbase[(size_t)row * (3 * C) + d];
    }
    for (int idx = tid; idx < 32 * 16; idx += 288) {   // zero pad rows 144..159
        int d = idx >> 4;
        vT[d * 160 + NTW + (idx & 15)] = (__bf16)0.0f;
    }
    __syncthreads();

    int t = wave;                // row strip 0..8 (9 waves x 16 rows = 144)

    // S = q @ k^T : A frag straight from global q, B frags from global k rows
    v16bf aq = load_a_frag(qbase, 3 * C, t * 16, 0, lane);
    v8f zc = {};
    v8f acc[9];
    for (int j = 0; j < 9; ++j) {
        v16bf bk = load_b_frag(kbase, 3 * C, j * 16, 0, lane);
        acc[j] = wmma_bf16(aq, bk, zc);
    }

    // Earth-position bias is affine: idx = qoff(row) + kb(col).  Hoist the
    // key-side decomposition (depends only on lane) out of the row loop.
    int kb[9];
    for (int j = 0; j < 9; ++j) {
        int nk = j * 16 + nl;
        int zk = nk / (WL * WW); int r2 = nk - zk * (WL * WW);
        int hk = r2 / WW, wk = r2 - hk * WW;
        kb[j] = (zk * (WP * WL * WL) + hk * WL) * (2 * WW - 1) - wk;
    }
    const float* bt = bias_table + (size_t)(u & (NW - 1)) * NH + hh;

    for (int v = 0; v < 8; ++v) {
        int nq = t * 16 + v + 8 * g;
        int zq = nq / (WL * WW); int r1 = nq - zq * (WL * WW);
        int hq = r1 / WW, wq = r1 - hq * WW;
        int qoff = (zq * (WL * WL) + hq) * (2 * WW - 1) + wq + (WW - 1);
        float rmax = -1e30f;
        for (int j = 0; j < 9; ++j) {
            float s = acc[j][v] + bt[(size_t)(qoff + kb[j]) * (NW * NH)];
            acc[j][v] = s;
            rmax = fmaxf(rmax, s);
        }
        for (int off = 8; off; off >>= 1) rmax = fmaxf(rmax, __shfl_xor(rmax, off, 32));
        float rsum = 0.f;
        for (int j = 0; j < 9; ++j) { float e = __expf(acc[j][v] - rmax); acc[j][v] = e; rsum += e; }
        for (int off = 8; off; off >>= 1) rsum += __shfl_xor(rsum, off, 32);
        float inv = 1.0f / rsum;
        int rloc = v + 8 * g;
        for (int j = 0; j < 9; ++j)
            Ps[t][rloc * 160 + j * 16 + nl] = f2bf(acc[j][v] * inv);
    }
    for (int i = lane; i < 16 * 16; i += 32)     // zero pad P cols 144..159
        Ps[t][(i >> 4) * 160 + NTW + (i & 15)] = (__bf16)0.0f;

    // O = P @ v   (K padded to 160, 5 WMMA steps x 2 N-tiles)
    v8f oa[2] = {};
    for (int kc = 0; kc < 160; kc += 32) {
        v16bf ap = load_a_frag(&Ps[t][0], 160, 0, kc, lane);
        oa[0] = wmma_bf16(ap, load_b_frag(vT, 160, 0, kc, lane), oa[0]);
        oa[1] = wmma_bf16(ap, load_b_frag(vT, 160, 16, kc, lane), oa[1]);
    }
    for (int nt = 0; nt < 2; ++nt)
        for (int v = 0; v < 8; ++v) {
            int row = u * NTW + t * 16 + v + 8 * g;
            int col = hh * HD + nt * 16 + nl;
            o[(size_t)row * C + col] = f2bf(oa[nt][v]);
        }
}

// ---------------- launcher --------------------------------------------------
extern "C" void kernel_launch(void* const* d_in, const int* in_sizes, int n_in,
                              void* d_out, int out_size, void* d_ws, size_t ws_size,
                              hipStream_t stream) {
    const float* x        = (const float*)d_in[0];
    const float* norm1_w  = (const float*)d_in[1];
    const float* norm1_b  = (const float*)d_in[2];
    const float* qkv_w    = (const float*)d_in[3];
    const float* qkv_b    = (const float*)d_in[4];
    const float* proj_w   = (const float*)d_in[5];
    const float* proj_b   = (const float*)d_in[6];
    const float* bias_tab = (const float*)d_in[7];
    const float* norm2_w  = (const float*)d_in[8];
    const float* norm2_b  = (const float*)d_in[9];
    const float* mlp_w1   = (const float*)d_in[10];
    const float* mlp_b1   = (const float*)d_in[11];
    const float* mlp_w2   = (const float*)d_in[12];
    const float* mlp_b2   = (const float*)d_in[13];

    char* p = (char*)d_ws;
    __bf16* h_win  = (__bf16*)p; p += (size_t)NTOK * C * 2;
    __bf16* qkvb   = (__bf16*)p; p += (size_t)NTOK * 3 * C * 2;
    __bf16* ob     = (__bf16*)p; p += (size_t)NTOK * C * 2;
    float*  x1     = (float*) p; p += (size_t)NTOK * C * 4;
    __bf16* h2     = (__bf16*)p; p += (size_t)NTOK * C * 2;
    __bf16* m1     = (__bf16*)p; p += (size_t)NTOK * 4 * C * 2;
    __bf16* qkv_wt = (__bf16*)p; p += (size_t)C * 3 * C * 2;
    __bf16* proj_wt= (__bf16*)p; p += (size_t)C * C * 2;
    __bf16* w1t    = (__bf16*)p; p += (size_t)C * 4 * C * 2;
    __bf16* w2t    = (__bf16*)p; p += (size_t)4 * C * C * 2;

    // weights -> bf16, transposed to [N][K]
    cvt_t_k<<<(C * 3 * C + 255) / 256, 256, 0, stream>>>(qkv_w, qkv_wt, C, 3 * C);
    cvt_t_k<<<(C * C + 255) / 256, 256, 0, stream>>>(proj_w, proj_wt, C, C);
    cvt_t_k<<<(C * 4 * C + 255) / 256, 256, 0, stream>>>(mlp_w1, w1t, C, 4 * C);
    cvt_t_k<<<(C * 4 * C + 255) / 256, 256, 0, stream>>>(mlp_w2, w2t, 4 * C, C);

    // LN1 + roll + window partition
    ln_k<<<NTOK, 256, 0, stream>>>(x, norm1_w, norm1_b, h_win, 1);

    // QKV projection
    gemm_k<0><<<dim3(3 * C / 256, NTOK / 128), 256, 0, stream>>>(
        h_win, qkv_wt, qkv_b, qkvb, nullptr, NTOK, 3 * C, C);

    // windowed attention
    attn_k<<<dim3(NWIN, NH), 288, 0, stream>>>(qkvb, bias_tab, ob);

    // output projection + un-window + residual  -> x1 (natural order, f32)
    gemm_k<1><<<dim3(C / 256, NTOK / 128), 256, 0, stream>>>(
        ob, proj_wt, proj_b, x1, x, NTOK, C, C);

    // LN2
    ln_k<<<NTOK, 256, 0, stream>>>(x1, norm2_w, norm2_b, h2, 0);

    // MLP
    gemm_k<2><<<dim3(4 * C / 256, NTOK / 128), 256, 0, stream>>>(
        h2, w1t, mlp_b1, m1, nullptr, NTOK, 4 * C, C);
    gemm_k<3><<<dim3(C / 256, NTOK / 128), 256, 0, stream>>>(
        m1, w2t, mlp_b2, (float*)d_out, x1, NTOK, C, 4 * C);
}
